// TimeSformerModel_13116830121993
// MI455X (gfx1250) — compile-verified, matrix operand order checked by reading, not run on hardware
//
#include <hip/hip_runtime.h>

// ---------------------------------------------------------------------------
// TimeSformer forward for MI455X (gfx1250): bf16 WMMA + async global->LDS DMA.
// ---------------------------------------------------------------------------

typedef __attribute__((ext_vector_type(16))) __bf16 v16bf;
typedef __attribute__((ext_vector_type(8)))  float  v8f;

#define S_TOK   1576        // T * N
#define D_MODEL 768
#define NHEAD   12
#define DHEAD   64
#define T_FR    8
#define N_TOK   197
#define NP_TOK  196
#define MLP_DIM 3072

#define BM 128
#define BN 64
#define BK 32

#define TS_USE_ASYNC_LDS 1

// fp32 -> bf16 round-to-nearest-even (bit manipulation; no scalar __bf16 math)
__device__ __forceinline__ unsigned short f2bf(float f) {
  unsigned u = __builtin_bit_cast(unsigned, f);
  u = (u + 0x7FFFu + ((u >> 16) & 1u)) >> 16;
  return (unsigned short)u;
}

#if TS_USE_ASYNC_LDS
// CDNA5 async DMA: copy 16 bytes global -> LDS, tracked by ASYNCcnt.
// LDS operand is the low 32 bits of the flat shared address (== LDS offset).
__device__ __forceinline__ void async_g2l_b128(unsigned short* lds_dst,
                                               const unsigned short* gsrc) {
  unsigned loff = (unsigned)(unsigned long long)lds_dst;
  asm volatile("global_load_async_to_lds_b128 %0, %1, off"
               :: "v"(loff), "v"(gsrc) : "memory");
}
__device__ __forceinline__ void async_wait0() {
  asm volatile("s_wait_asynccnt 0x0" ::: "memory");
}
#endif

union FragU { unsigned int u[8]; v16bf v; };

// Load one 16x32 (A) or 32x16 (B, stored transposed) bf16 fragment from LDS.
// p points at this lane's row (A: row m, B: column n); pairs (k,k+1) are
// contiguous, matching the CDNA5 16-bit WMMA VGPR layout.
__device__ __forceinline__ v16bf load_frag(const unsigned short* p, int lane) {
  FragU f;
  const int koff = (lane & 16) ? 8 : 0;
#pragma unroll
  for (int r = 0; r < 8; ++r) {
    const int k = (r < 4) ? (koff + 2 * r) : (16 + koff + 2 * (r - 4));
    f.u[r] = *(const unsigned int*)(p + k);
  }
  return f.v;
}

// ---------------------------------------------------------------------------
// Generic tiled GEMM:  out[M,N] = epilogue(A[M,K] @ B[K,N] + bias)
// flags: 1 = GELU (exact, erf), 2 = add residual (fp32), 4 = bf16 output
// ---------------------------------------------------------------------------
__global__ __launch_bounds__(256) void ts_gemm_bf16_kernel(
    const unsigned short* __restrict__ A, const unsigned short* __restrict__ Bw,
    const float* __restrict__ bias, const float* __restrict__ resid,
    float* __restrict__ outF, unsigned short* __restrict__ outB,
    int M, int N, int K, int flags) {
  __shared__ __align__(16) unsigned short lA[BM * BK];  // [row][k]
  __shared__ __align__(16) unsigned short lB[BN * BK];  // [n][k] (transposed)

  const int tid  = threadIdx.x;
  const int lane = tid & 31;
  const int wave = tid >> 5;
  const int wm   = wave & 3;   // 4 waves along M
  const int wn   = wave >> 2;  // 2 waves along N
  const int m0   = blockIdx.y * BM;
  const int n0   = blockIdx.x * BN;
  const int half = lane >> 4;
  const int nl   = lane & 15;

  // K-invariant A-tile copy descriptors: 512 b128 chunks, 2 per thread.
  const unsigned short* aSrc[2];
  unsigned short*       aDst[2];
#pragma unroll
  for (int i = 0; i < 2; ++i) {
    const int chunk = tid + 256 * i;       // 0..511
    const int row   = chunk >> 2;          // 4 x 16B per 64B row
    const int part  = chunk & 3;
    int gr = m0 + row; if (gr >= M) gr = M - 1;
    aSrc[i] = A + (size_t)gr * K + part * 8;
    aDst[i] = lA + row * BK + part * 8;
  }

  v8f acc[2][2] = {};

  for (int kk = 0; kk < K; kk += BK) {
    __syncthreads();   // previous iteration's fragment reads are complete
#if TS_USE_ASYNC_LDS
#pragma unroll
    for (int i = 0; i < 2; ++i) async_g2l_b128(aDst[i], aSrc[i] + kk);
#else
#pragma unroll
    for (int i = 0; i < 2; ++i) {
      FragU t; t.v = *(const v16bf*)(aSrc[i] + kk);
      *(v16bf*)aDst[i] = t.v;
    }
#endif
    // B tile (transposed into LDS): batch global loads, then LDS stores
    unsigned int rb[4];
    {
      const size_t koffB = (size_t)kk * N;
#pragma unroll
      for (int i = 0; i < 4; ++i) {
        const int kr = (tid >> 5) + 8 * i;
        rb[i] = *(const unsigned int*)(Bw + koffB + (size_t)kr * N + n0 + (tid & 31) * 2);
      }
    }
#pragma unroll
    for (int i = 0; i < 4; ++i) {
      const int kr = (tid >> 5) + 8 * i;
      lB[((tid & 31) * 2 + 0) * BK + kr] = (unsigned short)(rb[i] & 0xFFFFu);
      lB[((tid & 31) * 2 + 1) * BK + kr] = (unsigned short)(rb[i] >> 16);
    }
    // Prefetch next K-slab of B while this one computes
    if (kk + BK < K)
      __builtin_prefetch(Bw + (size_t)(kk + BK + (tid & 31)) * N + n0, 0, 1);
#if TS_USE_ASYNC_LDS
    async_wait0();
#endif
    __syncthreads();

    v16bf a0 = load_frag(lA + (wm * 32 +      nl) * BK, lane);
    v16bf a1 = load_frag(lA + (wm * 32 + 16 + nl) * BK, lane);
    v16bf b0 = load_frag(lB + (wn * 32 +      nl) * BK, lane);
    v16bf b1 = load_frag(lB + (wn * 32 + 16 + nl) * BK, lane);
    acc[0][0] = __builtin_amdgcn_wmma_f32_16x16x32_bf16(false, a0, false, b0, (short)0, acc[0][0], false, false);
    acc[0][1] = __builtin_amdgcn_wmma_f32_16x16x32_bf16(false, a0, false, b1, (short)0, acc[0][1], false, false);
    acc[1][0] = __builtin_amdgcn_wmma_f32_16x16x32_bf16(false, a1, false, b0, (short)0, acc[1][0], false, false);
    acc[1][1] = __builtin_amdgcn_wmma_f32_16x16x32_bf16(false, a1, false, b1, (short)0, acc[1][1], false, false);
  }

  // Epilogue. C layout: VGPR r -> row r (lanes 0-15) / r+8 (lanes 16-31)
#pragma unroll
  for (int fm = 0; fm < 2; ++fm)
#pragma unroll
    for (int fn = 0; fn < 2; ++fn) {
      const int nb = n0 + wn * 32 + fn * 16 + nl;
#pragma unroll
      for (int r = 0; r < 8; ++r) {
        const int m = m0 + wm * 32 + fm * 16 + r + 8 * half;
        if (m < M) {
          float v = acc[fm][fn][r] + bias[nb];
          if (flags & 1) v = 0.5f * v * (1.0f + erff(v * 0.70710678118f));
          if (flags & 2) v += resid[(size_t)m * N + nb];
          if (flags & 4) outB[(size_t)m * N + nb] = f2bf(v);
          else           outF[(size_t)m * N + nb] = v;
        }
      }
    }
}

// ---------------------------------------------------------------------------
// Flash-style attention. Q pre-scaled by DH^-0.5. Layout: [NH][S][64] bf16.
// grid = (ceil(L/64), nseg, NH); 4 waves/block, 16 query rows per wave.
// ---------------------------------------------------------------------------
__global__ __launch_bounds__(128) void ts_attn_flash_kernel(
    const unsigned short* __restrict__ Q, const unsigned short* __restrict__ Km,
    const unsigned short* __restrict__ Vm, float* __restrict__ O,
    int L, int nseg) {
  const int head = blockIdx.z;
  const int seg  = blockIdx.y;
  const int qt   = blockIdx.x;
  const int S    = L * nseg;
  const size_t hb = (size_t)head * S * DHEAD;
  const int s0   = seg * L;

  __shared__ __align__(16) unsigned short qT[64 * 64];       // [q][dh]
  __shared__ __align__(16) unsigned short kT[32 * 64];       // [key][dh]
  __shared__ __align__(16) unsigned short vT[64 * 32];       // [dh][key]
  __shared__ __align__(16) unsigned short pT[4 * 16 * 32];   // per-wave P tile

  const int tid  = threadIdx.x;
  const int lane = tid & 31;
  const int wave = tid >> 5;
  const int half = lane >> 4;
  const int nl   = lane & 15;

  // Stage 64x64 Q tile: 512 b128 chunks, 4 per thread (async DMA)
#pragma unroll
  for (int i = 0; i < 4; ++i) {
    const int chunk = tid + 128 * i;   // 0..511
    const int qr    = chunk >> 3;      // 8 x 16B per 128B row
    const int part  = chunk & 7;
    int gq = qt * 64 + qr; if (gq >= L) gq = L - 1;
#if TS_USE_ASYNC_LDS
    async_g2l_b128(qT + qr * 64 + part * 8,
                   Q + hb + (size_t)(s0 + gq) * DHEAD + part * 8);
#else
    FragU t; t.v = *(const v16bf*)(Q + hb + (size_t)(s0 + gq) * DHEAD + part * 8);
    *(v16bf*)(qT + qr * 64 + part * 8) = t.v;
#endif
  }
#if TS_USE_ASYNC_LDS
  async_wait0();
#endif
  __syncthreads();
  const v16bf aq0 = load_frag(qT + (wave * 16 + nl) * 64 +  0, lane);
  const v16bf aq1 = load_frag(qT + (wave * 16 + nl) * 64 + 32, lane);

  v8f accO[4] = {};
  float mi[8], li[8];
#pragma unroll
  for (int r = 0; r < 8; ++r) { mi[r] = -3.0e30f; li[r] = 0.0f; }

  for (int kb = 0; kb < L; kb += 32) {
    __syncthreads();
    // K block 32x64 (straight copy): 256 b128 chunks, 2 per thread (async DMA)
#pragma unroll
    for (int i = 0; i < 2; ++i) {
      const int chunk = tid + 128 * i;
      const int kr    = chunk >> 3;
      const int part  = chunk & 7;
      int gk = kb + kr; if (gk >= L) gk = L - 1;
#if TS_USE_ASYNC_LDS
      async_g2l_b128(kT + kr * 64 + part * 8,
                     Km + hb + (size_t)(s0 + gk) * DHEAD + part * 8);
#else
      FragU t; t.v = *(const v16bf*)(Km + hb + (size_t)(s0 + gk) * DHEAD + part * 8);
      *(v16bf*)(kT + kr * 64 + part * 8) = t.v;
#endif
    }
    // V block transposed -> [dh][key]: batch loads to regs, then LDS stores
    unsigned int rv[8];
#pragma unroll
    for (int i = 0; i < 8; ++i) {
      const int lin = tid + 128 * i;
      const int kr  = lin >> 5;
      int gk = kb + kr; if (gk >= L) gk = L - 1;
      rv[i] = *(const unsigned int*)(Vm + hb + (size_t)(s0 + gk) * DHEAD + (lin & 31) * 2);
    }
#pragma unroll
    for (int i = 0; i < 8; ++i) {
      const int lin = tid + 128 * i;
      const int kr  = lin >> 5;
      const int kd  = lin & 31;
      vT[(kd * 2 + 0) * 32 + kr] = (unsigned short)(rv[i] & 0xFFFFu);
      vT[(kd * 2 + 1) * 32 + kr] = (unsigned short)(rv[i] >> 16);
    }
#if TS_USE_ASYNC_LDS
    async_wait0();
#endif
    __syncthreads();

    // scores: two 16-key sub-tiles, each with two K=32 WMMA steps over dh
    v8f sc0 = {}; v8f sc1 = {};
    sc0 = __builtin_amdgcn_wmma_f32_16x16x32_bf16(false, aq0, false, load_frag(kT + (     nl) * 64 +  0, lane), (short)0, sc0, false, false);
    sc0 = __builtin_amdgcn_wmma_f32_16x16x32_bf16(false, aq1, false, load_frag(kT + (     nl) * 64 + 32, lane), (short)0, sc0, false, false);
    sc1 = __builtin_amdgcn_wmma_f32_16x16x32_bf16(false, aq0, false, load_frag(kT + (16 + nl) * 64 +  0, lane), (short)0, sc1, false, false);
    sc1 = __builtin_amdgcn_wmma_f32_16x16x32_bf16(false, aq1, false, load_frag(kT + (16 + nl) * 64 + 32, lane), (short)0, sc1, false, false);

    const bool inv0 = (kb + nl) >= L;
    const bool inv1 = (kb + 16 + nl) >= L;
#pragma unroll
    for (int r = 0; r < 8; ++r) {
      float x0 = inv0 ? -3.0e30f : sc0[r];
      float x1 = inv1 ? -3.0e30f : sc1[r];
      float sm = fmaxf(x0, x1);
#pragma unroll
      for (int msk = 1; msk < 16; msk <<= 1) sm = fmaxf(sm, __shfl_xor(sm, msk, 32));
      const float mn = fmaxf(mi[r], sm);
      const float c  = __expf(mi[r] - mn);
      const float p0 = __expf(x0 - mn);
      const float p1 = __expf(x1 - mn);
      float rs = p0 + p1;
#pragma unroll
      for (int msk = 1; msk < 16; msk <<= 1) rs += __shfl_xor(rs, msk, 32);
      li[r] = li[r] * c + rs;
      mi[r] = mn;
#pragma unroll
      for (int nt = 0; nt < 4; ++nt) accO[nt][r] *= c;
      const int prow = r + 8 * half;
      pT[wave * 512 + prow * 32 +      nl] = f2bf(p0);
      pT[wave * 512 + prow * 32 + 16 + nl] = f2bf(p1);
    }
    // P (16x32) @ V (32x64): 4 WMMAs across dh tiles
    const v16bf aP = load_frag(pT + wave * 512 + nl * 32, lane);
#pragma unroll
    for (int nt = 0; nt < 4; ++nt)
      accO[nt] = __builtin_amdgcn_wmma_f32_16x16x32_bf16(
          false, aP, false, load_frag(vT + (nt * 16 + nl) * 32, lane),
          (short)0, accO[nt], false, false);
  }

#pragma unroll
  for (int nt = 0; nt < 4; ++nt)
#pragma unroll
    for (int r = 0; r < 8; ++r) {
      const int m = qt * 64 + wave * 16 + r + 8 * half;
      if (m < L)
        O[(size_t)(s0 + m) * D_MODEL + head * DHEAD + nt * 16 + nl] =
            accO[nt][r] / li[r];
    }
}

// ---------------------------------------------------------------------------
// LayerNorm over D; bf16 output for GEMM feeds, fp32 for the final output.
// ---------------------------------------------------------------------------
__global__ __launch_bounds__(256) void ts_layernorm_kernel(
    const float* __restrict__ x, const float* __restrict__ g,
    const float* __restrict__ b, unsigned short* __restrict__ outB,
    float* __restrict__ outF, int D) {
  const int s = blockIdx.x;
  const float* row = x + (size_t)s * D;
  __shared__ float red[256];
  float sum = 0.f;
  for (int d = threadIdx.x; d < D; d += 256) sum += row[d];
  red[threadIdx.x] = sum; __syncthreads();
  for (int off = 128; off > 0; off >>= 1) {
    if (threadIdx.x < off) red[threadIdx.x] += red[threadIdx.x + off];
    __syncthreads();
  }
  const float mu = red[0] / D;
  __syncthreads();
  float vs = 0.f;
  for (int d = threadIdx.x; d < D; d += 256) { float t = row[d] - mu; vs += t * t; }
  red[threadIdx.x] = vs; __syncthreads();
  for (int off = 128; off > 0; off >>= 1) {
    if (threadIdx.x < off) red[threadIdx.x] += red[threadIdx.x + off];
    __syncthreads();
  }
  const float rstd = rsqrtf(red[0] / D + 1e-5f);
  for (int d = threadIdx.x; d < D; d += 256) {
    const float v = (row[d] - mu) * rstd * g[d] + b[d];
    if (outB) outB[(size_t)s * D + d] = f2bf(v);
    else      outF[(size_t)s * D + d] = v;
  }
}

// ------------------------- small elementwise kernels -----------------------
__global__ void ts_cvt_bf16_kernel(const float* __restrict__ in,
                                   unsigned short* __restrict__ out, int n) {
  int i = blockIdx.x * 256 + threadIdx.x;
  if (i < n) out[i] = f2bf(in[i]);
}

// W_patch is [D][K]; GEMM wants [K][N=D]
__global__ void ts_cvt_transpose_kernel(const float* __restrict__ in,
                                        unsigned short* __restrict__ out,
                                        int Nr, int Kc) {
  int i = blockIdx.x * 256 + threadIdx.x;
  if (i < Nr * Kc) {
    int n = i / Kc, k = i % Kc;
    out[(size_t)k * Nr + n] = f2bf(in[i]);
  }
}

// Extract per-frame 16x16 patches from x[1,T,3,224,224] -> bf16 [T*196][768]
__global__ void ts_patch_gather_kernel(const float* __restrict__ x,
                                       unsigned short* __restrict__ pmat) {
  int idx = blockIdx.x * 256 + threadIdx.x;
  const int total = T_FR * NP_TOK * D_MODEL;
  if (idx >= total) return;
  int row = idx / D_MODEL, col = idx % D_MODEL;
  int t = row / NP_TOK, i = row % NP_TOK;
  int c = col / 256, rem = col % 256, py = rem / 16, px = rem % 16;
  int iy = i / 14, ix = i % 14;
  size_t src = (((size_t)t * 3 + c) * 224 + iy * 16 + py) * 224 + ix * 16 + px;
  pmat[idx] = f2bf(x[src]);
}

// Apply torch's flatten/transpose axis scramble, prepend cls, add pos_embed.
__global__ void ts_assemble_kernel(const float* __restrict__ emb,
                                   const float* __restrict__ cls,
                                   const float* __restrict__ pos,
                                   float* __restrict__ h) {
  int idx = blockIdx.x * 256 + threadIdx.x;
  const int total = S_TOK * D_MODEL;
  if (idx >= total) return;
  int s = idx / D_MODEL, d = idx % D_MODEL;
  int t = s / N_TOK, j = s % N_TOK;
  float v;
  if (j == 0) v = cls[d];
  else {
    int flat = (j - 1) * D_MODEL + d;
    int ip = flat % NP_TOK, dp = flat / NP_TOK;
    v = emb[((size_t)t * NP_TOK + ip) * D_MODEL + dp];
  }
  h[idx] = v + pos[j * D_MODEL + d];
}

// Split fused qkv [S][3*768] into head-major bf16 Q/K/V; fold SCALE into Q.
__global__ void ts_qkv_split_kernel(const float* __restrict__ qkv,
                                    unsigned short* __restrict__ qb,
                                    unsigned short* __restrict__ kb,
                                    unsigned short* __restrict__ vb) {
  int idx = blockIdx.x * 256 + threadIdx.x;
  const int total = S_TOK * D_MODEL;
  if (idx >= total) return;
  int s = idx / D_MODEL, hd = idx % D_MODEL;
  int h = hd >> 6, d = hd & 63;
  size_t dst = ((size_t)h * S_TOK + s) * DHEAD + d;
  const float* p = qkv + (size_t)s * (3 * D_MODEL);
  qb[dst] = f2bf(p[hd] * 0.125f);            // DH^-0.5 = 1/8
  kb[dst] = f2bf(p[D_MODEL + hd]);
  vb[dst] = f2bf(p[2 * D_MODEL + hd]);
}

// Divided layers: temporal softmax over a length-1 axis is identity,
// so x_t == mean_n(v) per (head, frame).
__global__ __launch_bounds__(64) void ts_vmean_kernel(
    const float* __restrict__ qkv, float* __restrict__ vmean) {
  int h = blockIdx.x / T_FR, t = blockIdx.x % T_FR;
  int d = threadIdx.x;
  float s = 0.f;
  for (int n = 0; n < N_TOK; ++n)
    s += qkv[(size_t)(t * N_TOK + n) * (3 * D_MODEL) + 2 * D_MODEL + h * DHEAD + d];
  vmean[(h * T_FR + t) * DHEAD + d] = s * (1.0f / N_TOK);
}

__global__ void ts_attn_combine_kernel(const float* __restrict__ O,
                                       const float* __restrict__ vmean,
                                       unsigned short* __restrict__ obf,
                                       int add_vmean) {
  int idx = blockIdx.x * 256 + threadIdx.x;
  const int total = S_TOK * D_MODEL;
  if (idx >= total) return;
  int s = idx / D_MODEL, hd = idx % D_MODEL;
  float v = O[idx];
  if (add_vmean) {
    int h = hd >> 6, d = hd & 63, t = s / N_TOK;
    v += vmean[(h * T_FR + t) * DHEAD + d];
  }
  obf[idx] = f2bf(v);
}

// ---------------------------------------------------------------------------
extern "C" void kernel_launch(void* const* d_in, const int* in_sizes, int n_in,
                              void* d_out, int out_size, void* d_ws, size_t ws_size,
                              hipStream_t stream) {
  (void)in_sizes; (void)n_in; (void)out_size; (void)ws_size;
  const float* x       = (const float*)d_in[0];
  const float* W_patch = (const float*)d_in[1];
  const float* b_patch = (const float*)d_in[2];
  const float* cls     = (const float*)d_in[3];
  const float* pos     = (const float*)d_in[4];
  const float* ln1_g   = (const float*)d_in[5];
  const float* ln1_b   = (const float*)d_in[6];
  const float* qkv_w   = (const float*)d_in[7];
  const float* qkv_b   = (const float*)d_in[8];
  const float* proj_w  = (const float*)d_in[9];
  const float* proj_b  = (const float*)d_in[10];
  const float* ln2_g   = (const float*)d_in[11];
  const float* ln2_b   = (const float*)d_in[12];
  const float* fc1_w   = (const float*)d_in[13];
  const float* fc1_b   = (const float*)d_in[14];
  const float* fc2_w   = (const float*)d_in[15];
  const float* fc2_b   = (const float*)d_in[16];
  const float* lnf_g   = (const float*)d_in[17];
  const float* lnf_b   = (const float*)d_in[18];

  char* wp = (char*)d_ws;
  auto alloc = [&](size_t bytes) {
    char* p = wp; wp += (bytes + 255) & ~(size_t)255; return p;
  };
  const size_t SD = (size_t)S_TOK * D_MODEL;
  float*          h     = (float*)alloc(SD * 4);
  unsigned short* a_bf  = (unsigned short*)alloc(SD * 2);
  float*          qkv   = (float*)alloc(SD * 3 * 4);
  unsigned short* qb    = (unsigned short*)alloc(SD * 2);
  unsigned short* kb    = (unsigned short*)alloc(SD * 2);
  unsigned short* vb    = (unsigned short*)alloc(SD * 2);
  float*          Obuf  = (float*)alloc(SD * 4);
  unsigned short* o_bf  = (unsigned short*)alloc(SD * 2);
  unsigned short* m_bf  = (unsigned short*)alloc((size_t)S_TOK * MLP_DIM * 2);
  unsigned short* wbuf  = (unsigned short*)alloc((size_t)D_MODEL * MLP_DIM * 2);
  float*          vmean = (float*)alloc((size_t)NHEAD * T_FR * DHEAD * 4);
  // pre-layer phase reuses layer scratch:
  unsigned short* pmat = m_bf;                  // [1568][768] bf16
  float*          emb  = qkv;                   // [1568][768] fp32
  unsigned short* wpt  = wbuf;                  // [768][768] bf16 (W_patch^T)

  auto gemm = [&](const unsigned short* A, const unsigned short* Bw,
                  const float* bias, const float* resid, float* oF,
                  unsigned short* oB, int M, int N, int K, int flags) {
    dim3 grid((N + BN - 1) / BN, (M + BM - 1) / BM);
    ts_gemm_bf16_kernel<<<grid, 256, 0, stream>>>(A, Bw, bias, resid, oF, oB,
                                                  M, N, K, flags);
  };
  auto grid1 = [](int n) { return dim3((n + 255) / 256); };

  // ---- patch embedding -> token sequence ----
  ts_cvt_transpose_kernel<<<grid1(D_MODEL * D_MODEL), 256, 0, stream>>>(
      W_patch, wpt, D_MODEL, D_MODEL);
  ts_patch_gather_kernel<<<grid1(T_FR * NP_TOK * D_MODEL), 256, 0, stream>>>(x, pmat);
  gemm(pmat, wpt, b_patch, nullptr, emb, nullptr, T_FR * NP_TOK, D_MODEL, D_MODEL, 0);
  ts_assemble_kernel<<<grid1((int)SD), 256, 0, stream>>>(emb, cls, pos, h);

  // ---- transformer layers ----
  for (int i = 0; i < 12; ++i) {
    ts_layernorm_kernel<<<S_TOK, 256, 0, stream>>>(
        h, ln1_g + (size_t)i * D_MODEL, ln1_b + (size_t)i * D_MODEL, a_bf, nullptr, D_MODEL);
    ts_cvt_bf16_kernel<<<grid1(D_MODEL * 3 * D_MODEL), 256, 0, stream>>>(
        qkv_w + (size_t)i * D_MODEL * 3 * D_MODEL, wbuf, D_MODEL * 3 * D_MODEL);
    gemm(a_bf, wbuf, qkv_b + (size_t)i * 3 * D_MODEL, nullptr, qkv, nullptr,
         S_TOK, 3 * D_MODEL, D_MODEL, 0);
    ts_qkv_split_kernel<<<grid1((int)SD), 256, 0, stream>>>(qkv, qb, kb, vb);

    if ((i & 1) == 0) {  // divided space-time: per-frame spatial attn + v-mean
      ts_vmean_kernel<<<NHEAD * T_FR, 64, 0, stream>>>(qkv, vmean);
      ts_attn_flash_kernel<<<dim3((N_TOK + 63) / 64, T_FR, NHEAD), 128, 0, stream>>>(
          qb, kb, vb, Obuf, N_TOK, T_FR);
      ts_attn_combine_kernel<<<grid1((int)SD), 256, 0, stream>>>(Obuf, vmean, o_bf, 1);
    } else {             // joint space-time over all 1576 tokens
      ts_attn_flash_kernel<<<dim3((S_TOK + 63) / 64, 1, NHEAD), 128, 0, stream>>>(
          qb, kb, vb, Obuf, S_TOK, 1);
      ts_attn_combine_kernel<<<grid1((int)SD), 256, 0, stream>>>(Obuf, vmean, o_bf, 0);
    }

    ts_cvt_bf16_kernel<<<grid1(D_MODEL * D_MODEL), 256, 0, stream>>>(
        proj_w + (size_t)i * D_MODEL * D_MODEL, wbuf, D_MODEL * D_MODEL);
    gemm(o_bf, wbuf, proj_b + (size_t)i * D_MODEL, h, h, nullptr,
         S_TOK, D_MODEL, D_MODEL, 2);

    ts_layernorm_kernel<<<S_TOK, 256, 0, stream>>>(
        h, ln2_g + (size_t)i * D_MODEL, ln2_b + (size_t)i * D_MODEL, a_bf, nullptr, D_MODEL);
    ts_cvt_bf16_kernel<<<grid1(D_MODEL * MLP_DIM), 256, 0, stream>>>(
        fc1_w + (size_t)i * D_MODEL * MLP_DIM, wbuf, D_MODEL * MLP_DIM);
    gemm(a_bf, wbuf, fc1_b + (size_t)i * MLP_DIM, nullptr, nullptr, m_bf,
         S_TOK, MLP_DIM, D_MODEL, 1 | 4);
    ts_cvt_bf16_kernel<<<grid1(MLP_DIM * D_MODEL), 256, 0, stream>>>(
        fc2_w + (size_t)i * MLP_DIM * D_MODEL, wbuf, MLP_DIM * D_MODEL);
    gemm(m_bf, wbuf, fc2_b + (size_t)i * D_MODEL, h, h, nullptr,
         S_TOK, D_MODEL, MLP_DIM, 2);
  }

  // ---- final LayerNorm -> fp32 output [B,T,N,D] ----
  ts_layernorm_kernel<<<S_TOK, 256, 0, stream>>>(h, lnf_g, lnf_b, nullptr,
                                                 (float*)d_out, D_MODEL);
}
